// SingleHeadAttention_32229434589632
// MI455X (gfx1250) — matrix operand order repaired and checked by reading, not compiled, not run on hardware
//
#include <hip/hip_runtime.h>
#include <hip/hip_bf16.h>
#include <math.h>

// Problem constants (from reference)
constexpr int B = 8, S = 2048, H = 1024, D = 64;

typedef __attribute__((ext_vector_type(16))) __bf16 v16bf;
typedef __attribute__((ext_vector_type(8)))  __bf16 v8bf;
typedef __attribute__((ext_vector_type(8)))  float  v8f;

__device__ inline v16bf cat8(v8bf a, v8bf b) {
    v16bf r;
#pragma unroll
    for (int i = 0; i < 8; ++i) { r[i] = a[i]; r[i + 8] = b[i]; }
    return r;
}

__device__ inline float half_max16(float v) {
#pragma unroll
    for (int m = 1; m < 16; m <<= 1) v = fmaxf(v, __shfl_xor(v, m, 32));
    return v;
}
__device__ inline float half_sum16(float v) {
#pragma unroll
    for (int m = 1; m < 16; m <<= 1) v += __shfl_xor(v, m, 32);
    return v;
}

// Async 16-byte global -> LDS copy (per lane), tracked by ASYNCcnt.
// LDS address = low 32 bits of the flat pointer (shared->flat addrspacecast
// is {aperture_hi32, lds_addr32} on AMDGPU).
__device__ inline void async_ld16(void* ldsPtr, const void* gPtr) {
    unsigned int off = (unsigned int)(uintptr_t)ldsPtr;
    asm volatile("global_load_async_to_lds_b128 %0, %1, off"
                 :: "v"(off), "v"(gPtr)
                 : "memory");
}
__device__ inline void wait_async0() {
    asm volatile("s_wait_asynccnt 0x0" ::: "memory");
}

// ---------------------------------------------------------------------------
// Kernel 0: transpose + convert Wq/Wk/Wv [H][D] fp32 -> WtAll (3x [D][H]) bf16
// ---------------------------------------------------------------------------
__global__ void wt_kernel(const float* __restrict__ Wq,
                          const float* __restrict__ Wk,
                          const float* __restrict__ Wv,
                          __bf16* __restrict__ WtAll) {
    int idx = blockIdx.x * 256 + threadIdx.x;
    if (idx >= 3 * H * D) return;
    int m   = idx / (H * D);
    int rem = idx % (H * D);
    int h   = rem / D;
    int d   = rem % D;
    const float* W = (m == 0) ? Wq : (m == 1) ? Wk : Wv;
    WtAll[(size_t)m * D * H + (size_t)d * H + h] = (__bf16)W[(size_t)h * D + d];
}

// ---------------------------------------------------------------------------
// Kernel 1: fused QKV projection. One wave computes the 16-row x 64-col strip
// of ALL THREE of Q/K/V (x read & converted once, 12 WMMAs per A-fragment).
// All 12 B-fragments of a K-step are loaded first and pinned live by a single
// empty asm, forcing one 24-deep load clause + one wait + 12 WMMAs per step.
// Q,K stored bf16 [B][S][D]; V stored transposed bf16 [B][D][S].
// ---------------------------------------------------------------------------
__global__ void __launch_bounds__(128)
qkv_kernel(const float* __restrict__ x, const __bf16* __restrict__ WtAll,
           const float* __restrict__ bq, const float* __restrict__ bk,
           const float* __restrict__ bv,
           __bf16* __restrict__ Qb, __bf16* __restrict__ Kb,
           __bf16* __restrict__ Vt) {
    const int lane = threadIdx.x & 31;
    const int wave = threadIdx.x >> 5;
    const int lh   = lane & 15;   // 0..15
    const int hi   = lane >> 4;   // 0 or 1

    const int st = blockIdx.x * 4 + wave;    // seq tile, 0 .. B*S/16-1
    const int b  = st / (S / 16);
    const int s0 = (st % (S / 16)) * 16;

    const float* xp = x + ((size_t)b * S + s0) * H;

    v8f acc[3][4] = {};   // [matrix][16x16 C tile] -> 16x64 strips of Q,K,V

    for (int h0 = 0; h0 < H; h0 += 32) {
        // A fragment: x rows s0..s0+15, K-chunk h0..h0+31, fp32 -> bf16 once
        const float* ap = xp + (size_t)lh * H + h0 + hi * 8;
        v8f alo = *(const v8f*)ap;
        v8f ahi = *(const v8f*)(ap + 16);
        v16bf af;
#pragma unroll
        for (int i = 0; i < 8; ++i) {
            af[i]     = (__bf16)alo[i];
            af[i + 8] = (__bf16)ahi[i];
        }
        // Preload all 12 B fragments into distinct registers
        v16bf bfr[12];
#pragma unroll
        for (int m = 0; m < 3; ++m) {
            const __bf16* Wt = WtAll + (size_t)m * D * H;
#pragma unroll
            for (int j = 0; j < 4; ++j) {
                const __bf16* bp = Wt + (size_t)(16 * j + lh) * H + h0 + hi * 8;
                bfr[m * 4 + j] = cat8(*(const v8bf*)bp, *(const v8bf*)(bp + 16));
            }
        }
        // Pin all fragments live here: loads cannot be sunk past this point,
        // WMMAs cannot be hoisted above it -> one load clause, one wait,
        // then an uninterrupted 12-WMMA burst.
        asm volatile(""
                     : "+v"(bfr[0]), "+v"(bfr[1]), "+v"(bfr[2]), "+v"(bfr[3]),
                       "+v"(bfr[4]), "+v"(bfr[5]), "+v"(bfr[6]), "+v"(bfr[7]),
                       "+v"(bfr[8]), "+v"(bfr[9]), "+v"(bfr[10]),
                       "+v"(bfr[11]));
#pragma unroll
        for (int m = 0; m < 3; ++m)
#pragma unroll
            for (int j = 0; j < 4; ++j)
                acc[m][j] = __builtin_amdgcn_wmma_f32_16x16x32_bf16(
                    false, af, false, bfr[m * 4 + j], (short)0, acc[m][j],
                    false, false);
    }

    // bias add + store
#pragma unroll
    for (int m = 0; m < 3; ++m) {
        const float* bias = (m == 0) ? bq : (m == 1) ? bk : bv;
#pragma unroll
        for (int j = 0; j < 4; ++j) {
            const int   col = 16 * j + lh;
            const float bb  = bias[col];
#pragma unroll
            for (int r = 0; r < 8; ++r) {
                const int   srow = s0 + r + 8 * hi;
                const float val  = acc[m][j][r] + bb;
                if (m == 2) {
                    Vt[((size_t)b * D + col) * S + srow] = (__bf16)val;
                } else if (m == 0) {
                    Qb[((size_t)b * S + srow) * D + col] = (__bf16)val;
                } else {
                    Kb[((size_t)b * S + srow) * D + col] = (__bf16)val;
                }
            }
        }
    }
}

// ---------------------------------------------------------------------------
// Kernel 2: causal flash attention with async double-buffered K/V staging.
// 4 waves/block, each wave owns a 16-query tile (block = 64 query rows).
// K/V tiles for each 32-key block are cooperatively copied to LDS with
// global_load_async_to_lds_b128; next tile is prefetched during compute.
// ---------------------------------------------------------------------------
constexpr int KSTR = 80;  // LDS row stride (elems) for K tile rows of 64
constexpr int VSTR = 40;  // LDS row stride (elems) for V tile rows of 32

__global__ void __launch_bounds__(128)
attn_kernel(const __bf16* __restrict__ Qb, const __bf16* __restrict__ Kb,
            const __bf16* __restrict__ Vt, float* __restrict__ out) {
    __shared__ __align__(16) __bf16 kT[2][32][KSTR];  // 2 x 5 KB
    __shared__ __align__(16) __bf16 vT[2][64][VSTR];  // 2 x 5 KB
    __shared__ __align__(16) __bf16 pS[4][16][40];    // per-wave P staging

    const int tid  = threadIdx.x;
    const int lane = tid & 31;
    const int wave = tid >> 5;
    const int lh   = lane & 15;
    const int hi   = lane >> 4;

    const int b      = blockIdx.x / (S / 64);
    const int q0base = (blockIdx.x % (S / 64)) * 64;
    const int q0     = q0base + wave * 16;

    const __bf16* Qp = Qb + (size_t)b * S * D;
    const __bf16* Kp = Kb + (size_t)b * S * D;
    const __bf16* Vp = Vt + (size_t)b * D * S;

    // Q A-fragments (16 rows x D=64, split into two K=32 chunks)
    v16bf qf[2];
#pragma unroll
    for (int kk = 0; kk < 2; ++kk) {
        const __bf16* p = Qp + (size_t)(q0 + lh) * D + kk * 32 + hi * 8;
        qf[kk] = cat8(*(const v8bf*)p, *(const v8bf*)(p + 16));
    }

    v8f   acc[4] = {};
    float mrow[8], lrow[8];
#pragma unroll
    for (int r = 0; r < 8; ++r) { mrow[r] = -__builtin_inff(); lrow[r] = 0.0f; }

    // Block-uniform trip count: keys 0 .. q0base+63 in 32-key blocks.
    const int nb = q0base / 32 + 2;

    // Cooperative async staging of one 32-key K/V tile pair into buffer `buf`.
    auto stage_kv = [&](int t0, int buf) {
#pragma unroll
        for (int qq = 0; qq < 2; ++qq) {       // K: 256 chunks of 16 B
            int c = tid + qq * 128;
            int row = c >> 3, off = (c & 7) * 8;
            async_ld16(&kT[buf][row][off], Kp + (size_t)(t0 + row) * D + off);
        }
#pragma unroll
        for (int qq = 0; qq < 2; ++qq) {       // V^T: 256 chunks of 16 B
            int c = tid + qq * 128;
            int row = c >> 2, off = (c & 3) * 8;
            async_ld16(&vT[buf][row][off], Vp + (size_t)row * S + t0 + off);
        }
    };

    stage_kv(0, 0);   // prologue

    for (int kb = 0; kb < nb; ++kb) {
        const int t0  = kb * 32;
        const int buf = kb & 1;

        wait_async0();       // this wave's current-buffer loads landed in LDS
        __syncthreads();     // all waves' shares visible; prior buffer free

        if (kb + 1 < nb) stage_kv(t0 + 32, (kb + 1) & 1);  // prefetch

        if (t0 <= q0 + 15) {   // wave-uniform causal skip
            // ---- scores: S = Q @ K^T (two 16x16 tiles over 32 keys) ----
            v8f sc[2] = {};
#pragma unroll
            for (int n = 0; n < 2; ++n) {
#pragma unroll
                for (int kk = 0; kk < 2; ++kk) {
                    const __bf16* p = &kT[buf][16 * n + lh][kk * 32 + hi * 8];
                    v16bf kf = cat8(*(const v8bf*)p, *(const v8bf*)(p + 16));
                    sc[n] = __builtin_amdgcn_wmma_f32_16x16x32_bf16(
                        false, qf[kk], false, kf, (short)0, sc[n], false, false);
                }
            }

            // ---- causal mask ----
#pragma unroll
            for (int n = 0; n < 2; ++n) {
                const int t = t0 + 16 * n + lh;
#pragma unroll
                for (int r = 0; r < 8; ++r) {
                    const int q = q0 + r + 8 * hi;
                    if (t > q) sc[n][r] = -__builtin_inff();
                }
            }

            // ---- online softmax (row stats via half-wave shuffles) ----
            float mnew[8], scale[8];
#pragma unroll
            for (int r = 0; r < 8; ++r) {
                float bm = half_max16(fmaxf(sc[0][r], sc[1][r]));
                mnew[r]  = fmaxf(mrow[r], bm);
                scale[r] = __expf(mrow[r] - mnew[r]);
                mrow[r]  = mnew[r];
            }
            float p0[8], p1[8];
#pragma unroll
            for (int r = 0; r < 8; ++r) {
                p0[r] = __expf(sc[0][r] - mnew[r]);
                p1[r] = __expf(sc[1][r] - mnew[r]);
                lrow[r] = lrow[r] * scale[r] + half_sum16(p0[r] + p1[r]);
            }
#pragma unroll
            for (int j = 0; j < 4; ++j)
#pragma unroll
                for (int r = 0; r < 8; ++r) acc[j][r] *= scale[r];

            // ---- P: C layout -> LDS -> A fragment (per-wave, in-order DS) --
#pragma unroll
            for (int r = 0; r < 8; ++r) {
                pS[wave][r + 8 * hi][lh]      = (__bf16)p0[r];
                pS[wave][r + 8 * hi][16 + lh] = (__bf16)p1[r];
            }
            const __bf16* pr = &pS[wave][lh][hi * 8];
            v16bf pf = cat8(*(const v8bf*)pr, *(const v8bf*)(pr + 16));

            // ---- O += P @ V ----
#pragma unroll
            for (int j = 0; j < 4; ++j) {
                const __bf16* vp = &vT[buf][16 * j + lh][hi * 8];
                v16bf vf = cat8(*(const v8bf*)vp, *(const v8bf*)(vp + 16));
                acc[j] = __builtin_amdgcn_wmma_f32_16x16x32_bf16(
                    false, pf, false, vf, (short)0, acc[j], false, false);
            }
        }
    }

    // ---- epilogue: divide by l and by sqrt(H)=32, store fp32 ----
    float inv[8];
#pragma unroll
    for (int r = 0; r < 8; ++r) inv[r] = 1.0f / (lrow[r] * 32.0f);
#pragma unroll
    for (int j = 0; j < 4; ++j) {
        const int col = 16 * j + lh;
#pragma unroll
        for (int r = 0; r < 8; ++r) {
            const int q = q0 + r + 8 * hi;
            out[((size_t)b * S + q) * D + col] = acc[j][r] * inv[r];
        }
    }
}

// ---------------------------------------------------------------------------
extern "C" void kernel_launch(void* const* d_in, const int* in_sizes, int n_in,
                              void* d_out, int out_size, void* d_ws,
                              size_t ws_size, hipStream_t stream) {
    const float* x  = (const float*)d_in[0];
    const float* Wq = (const float*)d_in[1];
    const float* bq = (const float*)d_in[2];
    const float* Wk = (const float*)d_in[3];
    const float* bk = (const float*)d_in[4];
    const float* Wv = (const float*)d_in[5];
    const float* bv = (const float*)d_in[6];
    float* out = (float*)d_out;

    char* ws = (char*)d_ws;
    const size_t wtBytes  = (size_t)3 * D * H * sizeof(__bf16);   // 384 KB
    const size_t qkvBytes = (size_t)B * S * D * sizeof(__bf16);   // 2 MB each
    __bf16* WtAll = (__bf16*)(ws);
    __bf16* Qb    = (__bf16*)(ws + wtBytes);
    __bf16* Kb    = (__bf16*)(ws + wtBytes + qkvBytes);
    __bf16* Vt    = (__bf16*)(ws + wtBytes + 2 * qkvBytes);

    // 0) transpose+convert weights to bf16 [D][H]
    wt_kernel<<<(3 * H * D + 255) / 256, 256, 0, stream>>>(Wq, Wk, Wv, WtAll);

    // 1) fused QKV projection (x read once, pinned 24-load clause per K-step)
    qkv_kernel<<<B * (S / 16) / 4, 128, 0, stream>>>(
        x, WtAll, bq, bk, bv, Qb, Kb, Vt);

    // 2) causal flash attention (WMMA + async LDS double buffering)
    attn_kernel<<<B * (S / 64), 128, 0, stream>>>(Qb, Kb, Vt, out);

    (void)in_sizes; (void)n_in; (void)out_size; (void)ws_size;
}
